// TrsDecoderLayer_7791070675440
// MI455X (gfx1250) — matrix operand order repaired and checked by reading, not compiled, not run on hardware
//
#include <hip/hip_runtime.h>
#include <hip/hip_bf16.h>
#include <stdint.h>

// ---------------------------------------------------------------------------
// Types for CDNA5 WMMA (wave32)
// ---------------------------------------------------------------------------
typedef __attribute__((ext_vector_type(16))) _Float16 v16h;
typedef __attribute__((ext_vector_type(8)))  _Float16 v8h;
typedef __attribute__((ext_vector_type(8)))  float    v8f;

#define H_HEADS 8
#define F_DIM 256
#define E_FULL 272          // 8 heads * 34
#define KP 288              // 272 padded to multiple of 32
#define D_HEAD 34
#define DP 64               // head dim padded to multiple of 32
#define N_NODES 512
#define N_SIMS 64
#define M_ROWS (N_NODES * N_SIMS)   // 32768
#define SCALE_Q 0.17149858514250882f // 1/sqrt(34)
#define QKV_REG 320         // padded rows per q/k/v region (5 x 64)
#define QKV_PAD (3 * QKV_REG) // 960 padded Wp rows

// D = A(16x32 f16) * B(32x16 f16) + C(16x16 f32)
__device__ __forceinline__ v8f wmma16(v16h a, v16h b, v8f c) {
    return __builtin_amdgcn_wmma_f32_16x16x32_f16(
        /*neg_a=*/false, a, /*neg_b=*/false, b,
        /*c_mod=*/(short)0, c, /*reuse_a=*/false, /*reuse_b=*/false);
}

// A fragment, row-major f16 source, 16 rows starting at `base`, K offset k0.
// Per ISA 7.12.2: lanes 0-15 hold K = k0+[0..7, 16..23]; lanes 16-31 K = k0+[8..15, 24..31]
__device__ __forceinline__ v16h load_a_frag(const _Float16* __restrict__ base,
                                            int ld, int k0, int lane) {
    int row = lane & 15;
    int kb  = k0 + ((lane & 16) ? 8 : 0);
    const _Float16* p = base + (size_t)row * ld + kb;
    v8h lo = *(const v8h*)p;
    v8h hi = *(const v8h*)(p + 16);
    v16h a;
#pragma unroll
    for (int i = 0; i < 8; ++i) { a[i] = lo[i]; a[i + 8] = hi[i]; }
    return a;
}

// B fragment (32x16): N = lane&15; lanes 0-15 hold K = k0+[0..7, 16..23],
// lanes 16-31 K = k0+[8..15, 24..31] (matches A). Source rows are the N-columns.
__device__ __forceinline__ v16h load_b_frag(const _Float16* __restrict__ base,
                                            int ld, int k0, int lane) {
    int col = lane & 15;
    int kb  = k0 + ((lane & 16) ? 8 : 0);
    const _Float16* p = base + (size_t)col * ld + kb;
    v8h lo = *(const v8h*)p;
    v8h hi = *(const v8h*)(p + 16);
    v16h b;
#pragma unroll
    for (int i = 0; i < 8; ++i) { b[i] = lo[i]; b[i + 8] = hi[i]; }
    return b;
}

// ---------------------------------------------------------------------------
// Utility kernels
// ---------------------------------------------------------------------------
__global__ void zero_ws_kernel(uint4* p, size_t n16) {
    size_t i = (size_t)blockIdx.x * blockDim.x + threadIdx.x;
    size_t st = (size_t)gridDim.x * blockDim.x;
    uint4 z = {0u, 0u, 0u, 0u};
    for (; i < n16; i += st) p[i] = z;
}

// Build x_query / x_memo (group-concat of graph/memory with tiled prob).
__global__ void build_inputs_kernel(const float* __restrict__ prob,
                                    const float* __restrict__ graph,
                                    const float* __restrict__ memo,
                                    _Float16* __restrict__ xq16,
                                    _Float16* __restrict__ xm16,
                                    float* __restrict__ xq32) {
    int r = blockIdx.x;  // row = n*64 + s
    for (int j = threadIdx.x; j < E_FULL; j += blockDim.x) {
        int h = j / D_HEAD, t = j % D_HEAD;
        float vq, vm;
        if (t < 32) {
            vq = graph[(size_t)r * F_DIM + h * 32 + t];
            vm = memo [(size_t)r * F_DIM + h * 32 + t];
        } else {
            float pv = prob[(size_t)r * 2 + (t - 32)];
            vq = pv; vm = pv;
        }
        xq32[(size_t)r * E_FULL + j] = vq;
        xq16[(size_t)r * KP + j] = (_Float16)vq;
        xm16[(size_t)r * KP + j] = (_Float16)vm;
    }
}

// fp32 (rowsReal x K) -> f16 (rowsPad x Kp), zero padded rows/cols
__global__ void pack_w_kernel(const float* __restrict__ W, _Float16* __restrict__ Wh,
                              int rowsReal, int rowsPad, int K, int Kp) {
    int total = rowsPad * Kp;
    for (int idx = blockIdx.x * blockDim.x + threadIdx.x; idx < total;
         idx += gridDim.x * blockDim.x) {
        int rr = idx / Kp, cc = idx % Kp;
        Wh[idx] = (rr < rowsReal && cc < K) ? (_Float16)W[(size_t)rr * K + cc]
                                            : (_Float16)0.0f;
    }
}

// Wp (816 x 272) -> padded (960 x 288): 3 regions of 320 rows (272 real + 48 zero)
__global__ void pack_wp_kernel(const float* __restrict__ W, _Float16* __restrict__ Wh) {
    int total = QKV_PAD * KP;
    for (int idx = blockIdx.x * blockDim.x + threadIdx.x; idx < total;
         idx += gridDim.x * blockDim.x) {
        int p = idx / KP, cc = idx % KP;
        int type = p / QKV_REG, jj = p % QKV_REG;
        float v = 0.f;
        if (jj < E_FULL && cc < E_FULL)
            v = W[(size_t)(type * E_FULL + jj) * E_FULL + cc];
        Wh[idx] = (_Float16)v;
    }
}

// ---------------------------------------------------------------------------
// Fused QKV projection GEMM: (32768 x 288) x (960 x 288)^T, 16x64 per wave.
// Epilogue scatters q (scaled), k -> (b, node, DP), v -> transposed (b, DP, node).
// ---------------------------------------------------------------------------
__global__ __launch_bounds__(256)
void proj_qkv_kernel(const _Float16* __restrict__ Aq,
                     const _Float16* __restrict__ Akv,
                     const _Float16* __restrict__ Wp,   // 960 x 288 f16, padded
                     const float* __restrict__ bp,      // 816
                     _Float16* __restrict__ q_att,      // (512b, 512n, 64)
                     _Float16* __restrict__ k_att,      // (512b, 512n, 64)
                     _Float16* __restrict__ v_t) {      // (512b, 64, 512n)
    const int NTG = QKV_PAD / 64;  // 15 groups of 64 cols; 5 per type region
    int tid = threadIdx.x, lane = tid & 31;
    int gw = blockIdx.x * 8 + (tid >> 5);
    int mt = gw / NTG, ng = gw % NTG;
    if (mt >= (M_ROWS >> 4)) return;

    int type = ng / 5;                       // 0=q, 1=k, 2=v (group-uniform)
    const _Float16* A  = (type == 0 ? Aq : Akv) + (size_t)(mt << 4) * KP;
    const _Float16* Wb = Wp + (size_t)(ng << 6) * KP;

    v8f c[4] = {};
    for (int k0 = 0; k0 < KP; k0 += 32) {
        v16h a = load_a_frag(A, KP, k0, lane);     // loaded once, reused 4x
#pragma unroll
        for (int u = 0; u < 4; ++u)
            c[u] = wmma16(a, load_b_frag(Wb + (size_t)(u << 4) * KP, KP, k0, lane), c[u]);
    }

#pragma unroll
    for (int u = 0; u < 4; ++u) {
        int jp = (ng << 6) + (u << 4) + (lane & 15);   // padded col
        int jj = jp - type * QKV_REG;                  // col within region
        if (jj >= E_FULL) continue;                    // zero-padded column
        float bj = bp[type * E_FULL + jj];
        int h = jj / D_HEAD, dd = jj % D_HEAD;
#pragma unroll
        for (int r = 0; r < 8; ++r) {
            int i = (mt << 4) + r + ((lane & 16) ? 8 : 0);  // row = n*64 + s
            int n = i >> 6, s = i & 63;
            int b = (s << 3) + h;
            float v = c[u][r] + bj;
            if (type == 0)
                q_att[((size_t)b * N_NODES + n) * DP + dd] = (_Float16)(v * SCALE_Q);
            else if (type == 1)
                k_att[((size_t)b * N_NODES + n) * DP + dd] = (_Float16)v;
            else
                v_t[((size_t)b * DP + dd) * N_NODES + n] = (_Float16)v;
        }
    }
}

// ---------------------------------------------------------------------------
// Attention: per (batch b = s*8+h, 16-query tile). QK^T -> mask*scores ->
// softmax (fp32, LDS) -> f16 probs -> PV -> scatter into GEMM row layout.
// ---------------------------------------------------------------------------
__global__ __launch_bounds__(256)
void attn_kernel(const _Float16* __restrict__ q_att,
                 const _Float16* __restrict__ k_att,
                 const _Float16* __restrict__ v_t,
                 const float* __restrict__ mask,        // (512, 512)
                 _Float16* __restrict__ o_gemm) {       // (32768 x 288)
    __shared__ float    sc[16 * N_NODES];
    __shared__ _Float16 pr[16 * N_NODES];
    __shared__ float    red[16 * 16];
    __shared__ float    rowmax[16];
    __shared__ float    rowsum[16];

    int b  = blockIdx.y;   // s*8 + h
    int mt = blockIdx.x;   // query tile 0..31
    int tid = threadIdx.x, lane = tid & 31, w = tid >> 5;

    const _Float16* qb = q_att + (size_t)b * N_NODES * DP + (size_t)(mt << 4) * DP;
    const _Float16* kb = k_att + (size_t)b * N_NODES * DP;
    const _Float16* vb = v_t   + (size_t)b * DP * N_NODES;

    // Phase 1: scores = (q*scale) @ k^T, elementwise mask multiply
    v16h a0 = load_a_frag(qb, DP,  0, lane);
    v16h a1 = load_a_frag(qb, DP, 32, lane);
    for (int kt = w; kt < 32; kt += 8) {
        v8f c = {};
        c = wmma16(a0, load_b_frag(kb + (size_t)(kt << 4) * DP, DP,  0, lane), c);
        c = wmma16(a1, load_b_frag(kb + (size_t)(kt << 4) * DP, DP, 32, lane), c);
        int key = (kt << 4) + (lane & 15);
#pragma unroll
        for (int r = 0; r < 8; ++r) {
            int rl = r + ((lane & 16) ? 8 : 0);
            int qrow = (mt << 4) + rl;
            sc[rl * N_NODES + key] = c[r] * mask[(size_t)qrow * N_NODES + key];
        }
    }
    __syncthreads();

    // Softmax over 512 keys per row (16 threads per row, 32 cols each)
    {
        int row = tid >> 4, part = tid & 15;
        int c0 = part * 32;
        float m = -3.4e38f;
        for (int cjj = c0; cjj < c0 + 32; ++cjj) m = fmaxf(m, sc[row * N_NODES + cjj]);
        red[row * 16 + part] = m;
        __syncthreads();
        if (part == 0) {
            float mm = red[row * 16];
            for (int i = 1; i < 16; ++i) mm = fmaxf(mm, red[row * 16 + i]);
            rowmax[row] = mm;
        }
        __syncthreads();
        float mm = rowmax[row], ssum = 0.f;
        for (int cjj = c0; cjj < c0 + 32; ++cjj) {
            float e = __expf(sc[row * N_NODES + cjj] - mm);
            sc[row * N_NODES + cjj] = e;
            ssum += e;
        }
        red[row * 16 + part] = ssum;
        __syncthreads();
        if (part == 0) {
            float s2 = 0.f;
            for (int i = 0; i < 16; ++i) s2 += red[row * 16 + i];
            rowsum[row] = s2;
        }
        __syncthreads();
        float inv = 1.0f / rowsum[row];
        for (int cjj = c0; cjj < c0 + 32; ++cjj)
            pr[row * N_NODES + cjj] = (_Float16)(sc[row * N_NODES + cjj] * inv);
    }
    __syncthreads();

    // Phase 2: o = probs @ v  (waves 0..3, one 16-col tile of padded d each)
    if (w < 4) {
        int ct = w;
        v8f c = {};
        for (int kt = 0; kt < 16; ++kt) {           // K = 512 keys
            int k0 = kt << 5;
            int arow = lane & 15;
            int akb  = k0 + ((lane & 16) ? 8 : 0);
            v16h a;
#pragma unroll
            for (int e = 0; e < 8; ++e) {           // ds_load from shared probs
                a[e]     = pr[arow * N_NODES + akb + e];
                a[e + 8] = pr[arow * N_NODES + akb + 16 + e];
            }
            v16h bm = load_b_frag(vb + (size_t)(ct << 4) * N_NODES, N_NODES, k0, lane);
            c = wmma16(a, bm, c);
        }
        int dd = (ct << 4) + (lane & 15);
        if (dd < D_HEAD) {
            int s = b >> 3, h = b & 7;
#pragma unroll
            for (int r = 0; r < 8; ++r) {
                int n = (mt << 4) + r + ((lane & 16) ? 8 : 0);
                o_gemm[((size_t)n * N_SIMS + s) * KP + h * D_HEAD + dd] = (_Float16)c[r];
            }
        }
    }
}

// ---------------------------------------------------------------------------
// Generic f16 GEMM, 16x64 per wave: out32 = A @ W^T + bias (+ res),
// optional f16 mirror. M mult of 16, NoutPad mult of 64, Kp mult of 32.
// ---------------------------------------------------------------------------
__global__ __launch_bounds__(256)
void gemm_f16_kernel(const _Float16* __restrict__ A, int lda,
                     const _Float16* __restrict__ W, int ldw,  // NoutPad rows
                     const float* __restrict__ bias,           // NoutReal
                     const float* __restrict__ res,            // nullable, ld = ldo
                     float* __restrict__ out32, int ldo,
                     _Float16* __restrict__ out16, int ldo16,  // nullable
                     int M, int NoutPad, int NoutReal, int Kp) {
    int tid = threadIdx.x, lane = tid & 31;
    int gw = blockIdx.x * 8 + (tid >> 5);
    int ntg = NoutPad >> 6;
    int mt = gw / ntg, ng = gw % ntg;
    if (mt >= (M >> 4)) return;

    const _Float16* Ab = A + (size_t)(mt << 4) * lda;
    const _Float16* Wb = W + (size_t)(ng << 6) * ldw;

    v8f c[4] = {};
    for (int k0 = 0; k0 < Kp; k0 += 32) {
        v16h a = load_a_frag(Ab, lda, k0, lane);   // loaded once, reused 4x
#pragma unroll
        for (int u = 0; u < 4; ++u)
            c[u] = wmma16(a, load_b_frag(Wb + (size_t)(u << 4) * ldw, ldw, k0, lane), c[u]);
    }

    int rbase = (mt << 4) + ((lane & 16) ? 8 : 0);
#pragma unroll
    for (int u = 0; u < 4; ++u) {
        int col = (ng << 6) + (u << 4) + (lane & 15);
        if (col >= NoutReal) continue;             // padded column
        float bj = bias[col];
#pragma unroll
        for (int r = 0; r < 8; ++r) {
            int row = rbase + r;
            size_t o = (size_t)row * ldo + col;
            float v = c[u][r] + bj;
            if (res) v += res[o];
            out32[o] = v;
            if (out16) out16[(size_t)row * ldo16 + col] = (_Float16)v;
        }
    }
}

// ---------------------------------------------------------------------------
// Final split: t16 = tanh(y_f) in f16; prob head computed directly in fp32.
// ---------------------------------------------------------------------------
__global__ void final_small_kernel(const float* __restrict__ y,   // (32768 x 272)
                                   const float* __restrict__ Wy,  // (2 x 16)
                                   const float* __restrict__ by,  // (2)
                                   _Float16* __restrict__ t16,    // (32768 x 256)
                                   float* __restrict__ prob_out) {
    int r = blockIdx.x * blockDim.x + threadIdx.x;
    if (r >= M_ROWS) return;
    const float* yr = y + (size_t)r * E_FULL;
    float p[16];
#pragma unroll
    for (int h = 0; h < H_HEADS; ++h) {
        for (int t = 0; t < 32; ++t)
            t16[(size_t)r * F_DIM + h * 32 + t] = (_Float16)tanhf(yr[h * D_HEAD + t]);
        p[h * 2 + 0] = tanhf(yr[h * D_HEAD + 32]);
        p[h * 2 + 1] = tanhf(yr[h * D_HEAD + 33]);
    }
#pragma unroll
    for (int cc = 0; cc < 2; ++cc) {
        float a = by[cc];
        for (int t = 0; t < 16; ++t) a += p[t] * Wy[cc * 16 + t];
        prob_out[(size_t)r * 2 + cc] = a;
    }
}

// ---------------------------------------------------------------------------
// Host-side launch sequence
// ---------------------------------------------------------------------------
extern "C" void kernel_launch(void* const* d_in, const int* in_sizes, int n_in,
                              void* d_out, int out_size, void* d_ws, size_t ws_size,
                              hipStream_t stream) {
    (void)in_sizes; (void)n_in; (void)out_size;
    const float* seq_prob  = (const float*)d_in[0];
    const float* seq_graph = (const float*)d_in[1];
    const float* seq_mem   = (const float*)d_in[2];
    const float* mask      = (const float*)d_in[3];
    const float* Wp_self   = (const float*)d_in[4];
    const float* bp_self   = (const float*)d_in[5];
    const float* Wo_self   = (const float*)d_in[6];
    const float* bo_self   = (const float*)d_in[7];
    const float* Wp_cross  = (const float*)d_in[8];
    const float* bp_cross  = (const float*)d_in[9];
    const float* Wo_cross  = (const float*)d_in[10];
    const float* bo_cross  = (const float*)d_in[11];
    const float* Wf        = (const float*)d_in[12];
    const float* bf        = (const float*)d_in[13];
    const float* Wy        = (const float*)d_in[14];
    const float* by        = (const float*)d_in[15];

    // ---- workspace layout (bytes) ----
    const size_t SZ_X16  = (size_t)M_ROWS * KP * 2;        // 18,874,368
    const size_t SZ_ATT  = (size_t)512 * N_NODES * DP * 2; // 33,554,432
    const size_t SZ_X32  = (size_t)M_ROWS * E_FULL * 4;    // 35,651,584
    const size_t SZ_T16  = (size_t)M_ROWS * F_DIM * 2;     // 16,777,216

    size_t off = 0;
    // zero-initialized region (f16 buffers with padded columns/d)
    const size_t o_XQ16 = off; off += SZ_X16;
    const size_t o_XM16 = off; off += SZ_X16;
    const size_t o_X16  = off; off += SZ_X16;
    const size_t o_OG   = off; off += SZ_X16;
    const size_t o_QA   = off; off += SZ_ATT;
    const size_t zero_bytes = off;
    // rest
    const size_t o_KA   = off; off += SZ_ATT;
    const size_t o_VT   = off; off += SZ_ATT;
    const size_t o_T16  = off; off += SZ_T16;
    const size_t o_XQ32 = off; off += SZ_X32;   // also reused as y_f32
    const size_t o_X32  = off; off += SZ_X32;
    const size_t o_WPS  = off; off += (size_t)QKV_PAD * KP * 2;  // 552,960
    const size_t o_WPC  = off; off += (size_t)QKV_PAD * KP * 2;
    const size_t o_WOS  = off; off += (size_t)320 * KP * 2;      // 184,320
    const size_t o_WOC  = off; off += (size_t)320 * KP * 2;
    const size_t o_WF   = off; off += (size_t)F_DIM * F_DIM * 2;
    if (ws_size < off) return;  // insufficient scratch

    char* ws = (char*)d_ws;
    _Float16* XQ16 = (_Float16*)(ws + o_XQ16);
    _Float16* XM16 = (_Float16*)(ws + o_XM16);
    _Float16* X16  = (_Float16*)(ws + o_X16);
    _Float16* OG   = (_Float16*)(ws + o_OG);
    _Float16* QA   = (_Float16*)(ws + o_QA);
    _Float16* KA   = (_Float16*)(ws + o_KA);
    _Float16* VT   = (_Float16*)(ws + o_VT);
    _Float16* T16  = (_Float16*)(ws + o_T16);
    float*    XQ32 = (float*)(ws + o_XQ32);
    float*    Y32  = XQ32;   // reuse after x_query residual is consumed
    float*    X32  = (float*)(ws + o_X32);
    _Float16* WPS  = (_Float16*)(ws + o_WPS);
    _Float16* WPC  = (_Float16*)(ws + o_WPC);
    _Float16* WOS  = (_Float16*)(ws + o_WOS);
    _Float16* WOC  = (_Float16*)(ws + o_WOC);
    _Float16* WFH  = (_Float16*)(ws + o_WF);

    float* prob_out = (float*)d_out;                 // (32768 x 2)
    float* mem_out  = (float*)d_out + (size_t)M_ROWS * 2;  // (32768 x 256)

    // 1. zero padded f16 buffers
    zero_ws_kernel<<<2048, 256, 0, stream>>>((uint4*)ws, zero_bytes / 16);

    // 2. build x_query / x_memo
    build_inputs_kernel<<<M_ROWS, 256, 0, stream>>>(seq_prob, seq_graph, seq_mem,
                                                    XQ16, XM16, XQ32);

    // 3. pack weights to f16 (rows padded to group multiples, K padded)
    pack_wp_kernel<<<512, 256, 0, stream>>>(Wp_self,  WPS);
    pack_wp_kernel<<<512, 256, 0, stream>>>(Wp_cross, WPC);
    pack_w_kernel<<<256, 256, 0, stream>>>(Wo_self,  WOS, E_FULL, 320, E_FULL, KP);
    pack_w_kernel<<<256, 256, 0, stream>>>(Wo_cross, WOC, E_FULL, 320, E_FULL, KP);
    pack_w_kernel<<<256, 256, 0, stream>>>(Wf,       WFH, F_DIM, F_DIM, F_DIM, F_DIM);

    const int projBlocks = (2048 * (QKV_PAD / 64)) / 8;   // 2048*15/8 = 3840
    const int outpBlocks = (2048 * (320 / 64)) / 8;       // 2048*5/8  = 1280
    const int finBlocks  = (2048 * (F_DIM / 64)) / 8;     // 2048*4/8  = 1024
    dim3 attnGrid(32, 512);

    // ---- self attention ----
    proj_qkv_kernel<<<projBlocks, 256, 0, stream>>>(XQ16, XQ16, WPS, bp_self, QA, KA, VT);
    attn_kernel<<<attnGrid, 256, 0, stream>>>(QA, KA, VT, mask, OG);
    gemm_f16_kernel<<<outpBlocks, 256, 0, stream>>>(OG, KP, WOS, KP, bo_self,
                                                    XQ32, X32, E_FULL, X16, KP,
                                                    M_ROWS, 320, E_FULL, KP);
    // ---- cross attention (q from x, k/v from x_memo) ----
    proj_qkv_kernel<<<projBlocks, 256, 0, stream>>>(X16, XM16, WPC, bp_cross, QA, KA, VT);
    attn_kernel<<<attnGrid, 256, 0, stream>>>(QA, KA, VT, mask, OG);
    gemm_f16_kernel<<<outpBlocks, 256, 0, stream>>>(OG, KP, WOC, KP, bo_cross,
                                                    X32, Y32, E_FULL, (_Float16*)nullptr, 0,
                                                    M_ROWS, 320, E_FULL, KP);
    // ---- final heads ----
    final_small_kernel<<<M_ROWS / 256, 256, 0, stream>>>(Y32, Wy, by, T16, prob_out);
    gemm_f16_kernel<<<finBlocks, 256, 0, stream>>>(T16, F_DIM, WFH, F_DIM, bf,
                                                   (const float*)nullptr, mem_out, F_DIM,
                                                   (_Float16*)nullptr, 0,
                                                   M_ROWS, F_DIM, F_DIM, F_DIM);
}
